// Attention_57690000720083
// MI455X (gfx1250) — compile-verified, matrix operand order checked by reading, not run on hardware
//
#include <hip/hip_runtime.h>
#include <stdint.h>

typedef __attribute__((ext_vector_type(16))) __bf16 v16bf;
typedef __attribute__((ext_vector_type(8)))  float  v8f;
typedef __attribute__((ext_vector_type(4)))  unsigned int u32x4;
typedef __attribute__((ext_vector_type(8)))  int  i32x8;
typedef __attribute__((ext_vector_type(4)))  int  i32x4;

#define DIM      1024
#define HEADS    16
#define HEAD_DIM 64
#define SEQ      2048
#define BATCH    4
#define NROWS    (BATCH * SEQ)        // 8192
#define QK_SCALE 0.03125f             // 1024^-0.5 (reference uses full dim)

// workspace layout in bf16 elements (4 x 16MB = 64MB)
#define Q_OFF   ((size_t)0)
#define K_OFF   ((size_t)NROWS * DIM)         // 8388608
#define VT_OFF  ((size_t)2 * NROWS * DIM)
#define XO_OFF  ((size_t)3 * NROWS * DIM)

union Frag { v16bf v; uint32_t u[8]; };

__device__ __forceinline__ int frag_k(int j, int group) {
    // 16-bit A/B fragment K mapping (ISA 7.12.2): lanes 0-15 get K {0..7,16..23},
    // lanes 16-31 get K {8..15,24..31}; VGPR j holds a consecutive K pair.
    return ((j & 3) << 1) + (group << 3) + ((j >> 2) << 4);
}

// single-instruction f32->bf16 pair pack: v_perm_b32 selects hi[3:2]||lo[3:2]
__device__ __forceinline__ uint32_t pack_bf16(float a, float b) {
    union { float f; uint32_t u; } ua, ub;
    ua.f = a; ub.f = b;
    return __builtin_amdgcn_perm(ub.u, ua.u, 0x07060302u);
}

// ---------------------------------------------------------------------------
// TDM: issue one tensor_load_to_lds for a [rows x 32] bf16 tile with row
// stride DIM, from `gaddr` into LDS byte offset `lds_addr`. Wave-level op
// (EXEC ignored); caller guards to one wave and waits TENSORcnt.
// ---------------------------------------------------------------------------
__device__ __forceinline__ void tdm_load_tile_bf16_32xN(uint64_t gaddr, uint32_t lds_addr,
                                                        int rows) {
    u32x4 g0;
    g0[0] = 1u;                                            // count=1, user mode
    g0[1] = lds_addr;                                      // LDS byte address
    g0[2] = (uint32_t)gaddr;                               // global addr [31:0]
    g0[3] = ((uint32_t)(gaddr >> 32) & 0x01FFFFFFu) | (2u << 30); // [56:32] | type=2

    i32x8 g1;
    g1[0] = 1 << 16;                 // workgroup_mask=0, data_size=1 (2 bytes)
    g1[1] = 32 << 16;                // atomic_barrier_addr=0 | tensor_dim0=32
    g1[2] = rows << 16;              // tensor_dim0[31:16]=0 | tensor_dim1=rows
    g1[3] = 32 << 16;                // tensor_dim1[31:16]=0 | tile_dim0=32
    g1[4] = rows;                    // tile_dim1=rows, tile_dim2=0
    g1[5] = DIM;                     // tensor_dim0_stride[31:0] = 1024 elements
    g1[6] = 0;                       // stride0[47:32]=0 | stride1[15:0]=0
    g1[7] = 0;                       // stride1[47:16]=0

    i32x4 gz = {0, 0, 0, 0};
#if __clang_major__ >= 23
    i32x8 gz8 = {0, 0, 0, 0, 0, 0, 0, 0};
    __builtin_amdgcn_tensor_load_to_lds(g0, g1, gz, gz, gz8, 0);
#else
    __builtin_amdgcn_tensor_load_to_lds(g0, g1, gz, gz, 0);
#endif
}

// ---------------------------------------------------------------------------
// Kernel 1: qkv = x @ w_qkv^T   (M=8192, K=1024, N=3072), scatter epilogue
// block: 256 thr = 8 waves; tile 64(M) x 128(N); wave = 32x32 (2x2 wmma)
// Each 128-wide N tile sits inside exactly one of the q/k/v 1024-col
// segments, so the scatter target is block-uniform (no per-lane branching).
// ---------------------------------------------------------------------------
__global__ __launch_bounds__(256)
void qkv_gemm_kernel(const float* __restrict__ x, const float* __restrict__ w,
                     __bf16* __restrict__ q, __bf16* __restrict__ kk,
                     __bf16* __restrict__ vt)
{
    __shared__ __bf16 sX[64 * 32];
    __shared__ __bf16 sW[128 * 32];
    uint32_t* sX32 = (uint32_t*)sX;
    uint32_t* sW32 = (uint32_t*)sW;

    const int t     = threadIdx.x;
    const int lane  = t & 31;
    const int wave  = t >> 5;
    const int group = lane >> 4;
    const int laneN = lane & 15;
    const int m0    = blockIdx.y * 64;
    const int e0    = blockIdx.x * 128;
    const int wm    = wave >> 2;   // 0..1
    const int wn    = wave & 3;    // 0..3

    v8f acc[2][2];
#pragma unroll
    for (int i = 0; i < 2; ++i)
#pragma unroll
        for (int j = 0; j < 2; ++j) acc[i][j] = (v8f){};

    const int xrow = t >> 2, xseg = t & 3;

#pragma unroll 2
    for (int kt = 0; kt < DIM; kt += 32) {
        { // stage X tile 64x32 (fp32 -> bf16)
            const float4* g = (const float4*)(x + (size_t)(m0 + xrow) * DIM + kt + xseg * 8);
            float4 a = g[0], b = g[1];
            int o = (xrow * 32 + xseg * 8) >> 1;
            sX32[o + 0] = pack_bf16(a.x, a.y);
            sX32[o + 1] = pack_bf16(a.z, a.w);
            sX32[o + 2] = pack_bf16(b.x, b.y);
            sX32[o + 3] = pack_bf16(b.z, b.w);
        }
#pragma unroll
        for (int s = 0; s < 2; ++s) { // stage W tile 128x32
            int task = t + s * 256;
            int wrow = task >> 2, wseg = task & 3;
            const float4* g = (const float4*)(w + (size_t)(e0 + wrow) * DIM + kt + wseg * 8);
            float4 a = g[0], b = g[1];
            int o = (wrow * 32 + wseg * 8) >> 1;
            sW32[o + 0] = pack_bf16(a.x, a.y);
            sW32[o + 1] = pack_bf16(a.z, a.w);
            sW32[o + 2] = pack_bf16(b.x, b.y);
            sW32[o + 3] = pack_bf16(b.z, b.w);
        }
        __syncthreads();

        Frag af[2], bf[2];
#pragma unroll
        for (int s = 0; s < 2; ++s) {
            int arow = wm * 32 + s * 16 + laneN;
            int brow = wn * 32 + s * 16 + laneN;
#pragma unroll
            for (int j = 0; j < 8; ++j) {
                int k = frag_k(j, group);
                af[s].u[j] = sX32[arow * 16 + (k >> 1)];
                bf[s].u[j] = sW32[brow * 16 + (k >> 1)];
            }
        }
#pragma unroll
        for (int i = 0; i < 2; ++i)
#pragma unroll
            for (int j = 0; j < 2; ++j)
                acc[i][j] = __builtin_amdgcn_wmma_f32_16x16x32_bf16(
                    false, af[i].v, false, bf[j].v, (short)0, acc[i][j], false, false);
        __syncthreads();
    }

    // uniform scatter: whole block writes exactly one of q / k / v^T
    const int which = e0 >> 10;
#pragma unroll
    for (int i = 0; i < 2; ++i)
#pragma unroll
        for (int j = 0; j < 2; ++j) {
            const int mBase = m0 + wm * 32 + i * 16 + (group << 3); // rows mBase..mBase+7
            const int e     = e0 + wn * 32 + j * 16 + laneN;
            const int rem   = e & 1023;
            const int h     = rem >> 6;
            const int d     = rem & 63;
            const int b     = mBase >> 11;          // same batch for all 8 rows
            const int bh    = b * HEADS + h;
            if (which == 0) {
#pragma unroll
                for (int r = 0; r < 8; ++r) {
                    int n = (mBase + r) & 2047;
                    q[((size_t)bh * SEQ + n) * HEAD_DIM + d] =
                        (__bf16)(acc[i][j][r] * QK_SCALE);
                }
            } else if (which == 1) {
#pragma unroll
                for (int r = 0; r < 8; ++r) {
                    int n = (mBase + r) & 2047;
                    kk[((size_t)bh * SEQ + n) * HEAD_DIM + d] = (__bf16)acc[i][j][r];
                }
            } else {
                // v^T: the 8 accumulator rows are consecutive in n -> one 16B store
                int n0 = mBase & 2047;
                uint4 pk;
                pk.x = pack_bf16(acc[i][j][0], acc[i][j][1]);
                pk.y = pack_bf16(acc[i][j][2], acc[i][j][3]);
                pk.z = pack_bf16(acc[i][j][4], acc[i][j][5]);
                pk.w = pack_bf16(acc[i][j][6], acc[i][j][7]);
                *(uint4*)(vt + ((size_t)bh * HEAD_DIM + d) * SEQ + n0) = pk;
            }
        }
}

// ---------------------------------------------------------------------------
// Kernel 2: flash attention per (b,h); one wave = 16 query rows, online
// softmax over 32-key blocks. K/V^T stream from global (L2-resident).
// ---------------------------------------------------------------------------
__global__ __launch_bounds__(256)
void attn_kernel(const __bf16* __restrict__ q, const __bf16* __restrict__ kk,
                 const __bf16* __restrict__ vt, __bf16* __restrict__ xo)
{
    __shared__ __bf16 sP[8][16 * 32];   // per-wave score tile (C-layout -> A-layout)

    const int t     = threadIdx.x;
    const int lane  = t & 31;
    const int wave  = t >> 5;
    const int group = lane >> 4;
    const int laneN = lane & 15;

    const int bh    = blockIdx.x >> 4;      // 0..63
    const int rblk  = blockIdx.x & 15;
    const int qrow0 = rblk * 128 + wave * 16;
    const int b     = bh >> 4;
    const int h     = bh & 15;

    const uint32_t* q32 = (const uint32_t*)q;
    const uint32_t* k32 = (const uint32_t*)kk;
    const uint32_t* v32 = (const uint32_t*)vt;
    uint32_t* sP32 = (uint32_t*)sP[wave];

    // Q fragments: 2 K-steps covering head_dim 64 (resident whole kernel)
    Frag qa[2];
#pragma unroll
    for (int s = 0; s < 2; ++s) {
        int n = qrow0 + laneN;
        size_t base = ((size_t)bh * SEQ + n) * HEAD_DIM;
#pragma unroll
        for (int j = 0; j < 8; ++j)
            qa[s].u[j] = q32[(base + 32 * s + frag_k(j, group)) >> 1];
    }

    v8f acc[4];
#pragma unroll
    for (int j = 0; j < 4; ++j) acc[j] = (v8f){};
    float rmax[8], rsum[8];
#pragma unroll
    for (int r = 0; r < 8; ++r) { rmax[r] = -3.0e38f; rsum[r] = 0.0f; }

    for (int kb = 0; kb < SEQ; kb += 32) {
        // scores: two 16-wide key tiles, K-loop over head_dim (2 x wmma each)
        v8f sc[2];
#pragma unroll
        for (int tt = 0; tt < 2; ++tt) {
            v8f s0 = (v8f){};
            int key = kb + tt * 16 + laneN;
            size_t base = ((size_t)bh * SEQ + key) * HEAD_DIM;
#pragma unroll
            for (int s = 0; s < 2; ++s) {
                Frag bf;
#pragma unroll
                for (int j = 0; j < 8; ++j)
                    bf.u[j] = k32[(base + 32 * s + frag_k(j, group)) >> 1];
                s0 = __builtin_amdgcn_wmma_f32_16x16x32_bf16(
                    false, qa[s].v, false, bf.v, (short)0, s0, false, false);
            }
            sc[tt] = s0;
        }

        // online softmax update; row m = r + 8*group lives across a 16-lane half
#pragma unroll
        for (int r = 0; r < 8; ++r) {
            float mb = fmaxf(sc[0][r], sc[1][r]);
#pragma unroll
            for (int off = 1; off < 16; off <<= 1)
                mb = fmaxf(mb, __shfl_xor(mb, off, 32));
            float nm   = fmaxf(rmax[r], mb);
            float corr = __expf(rmax[r] - nm);
            rmax[r]    = nm;
            float p0   = __expf(sc[0][r] - nm);
            float p1   = __expf(sc[1][r] - nm);
            float ps   = p0 + p1;
#pragma unroll
            for (int off = 1; off < 16; off <<= 1)
                ps += __shfl_xor(ps, off, 32);
            rsum[r] = rsum[r] * corr + ps;
#pragma unroll
            for (int j = 0; j < 4; ++j) acc[j][r] *= corr;
            int row = r + (group << 3);
            sP[wave][row * 32 + laneN]      = (__bf16)p0;
            sP[wave][row * 32 + 16 + laneN] = (__bf16)p1;
        }
        __builtin_amdgcn_wave_barrier();
        asm volatile("s_wait_dscnt 0x0" ::: "memory");

        // re-read probabilities in A-fragment layout
        Frag pa;
#pragma unroll
        for (int j = 0; j < 8; ++j)
            pa.u[j] = sP32[laneN * 16 + (frag_k(j, group) >> 1)];
        __builtin_amdgcn_wave_barrier();

        // PV accumulate: 4 output column tiles over head_dim
#pragma unroll
        for (int jt = 0; jt < 4; ++jt) {
            Frag bv;
            int d = jt * 16 + laneN;
            size_t base = ((size_t)bh * HEAD_DIM + d) * SEQ + kb;
#pragma unroll
            for (int j = 0; j < 8; ++j)
                bv.u[j] = v32[(base + frag_k(j, group)) >> 1];
            acc[jt] = __builtin_amdgcn_wmma_f32_16x16x32_bf16(
                false, pa.v, false, bv.v, (short)0, acc[jt], false, false);
        }
    }

    // normalize and write attention output as [b*n, h*64+d] bf16
#pragma unroll
    for (int r = 0; r < 8; ++r) {
        float inv = 1.0f / rsum[r];
        int m    = r + (group << 3);
        int rowg = b * SEQ + qrow0 + m;
#pragma unroll
        for (int jt = 0; jt < 4; ++jt) {
            int col = h * HEAD_DIM + jt * 16 + laneN;
            xo[(size_t)rowg * DIM + col] = (__bf16)(acc[jt][r] * inv);
        }
    }
}

// ---------------------------------------------------------------------------
// Kernel 3: out = xo @ w_out^T + b_out   (M=8192, K=1024, N=1024), fp32 out
// X tile (already bf16) staged by the Tensor Data Mover; W staged by threads
// (needs fp32->bf16 conversion).
// ---------------------------------------------------------------------------
__global__ __launch_bounds__(256)
void out_gemm_kernel(const __bf16* __restrict__ xo, const float* __restrict__ w,
                     const float* __restrict__ bias, float* __restrict__ out)
{
    __shared__ __bf16 sX[64 * 32];
    __shared__ __bf16 sW[128 * 32];
    uint32_t* sX32 = (uint32_t*)sX;
    uint32_t* sW32 = (uint32_t*)sW;

    const int t     = threadIdx.x;
    const int lane  = t & 31;
    const int wave  = t >> 5;
    const int group = lane >> 4;
    const int laneN = lane & 15;
    const int m0    = blockIdx.y * 64;
    const int e0    = blockIdx.x * 128;
    const int wm    = wave >> 2;
    const int wn    = wave & 3;

    const uint32_t sX_lds = (uint32_t)(uintptr_t)(void*)sX;  // LDS byte offset

    v8f acc[2][2];
#pragma unroll
    for (int i = 0; i < 2; ++i)
#pragma unroll
        for (int j = 0; j < 2; ++j) acc[i][j] = (v8f){};

#pragma unroll 2
    for (int kt = 0; kt < DIM; kt += 32) {
        // TDM: DMA the 64x32 bf16 X tile straight into LDS (wave-level op)
        if (wave == 0) {
            uint64_t gaddr = (uint64_t)(uintptr_t)(xo + (size_t)m0 * DIM + kt);
            tdm_load_tile_bf16_32xN(gaddr, sX_lds, 64);
        }
#pragma unroll
        for (int s = 0; s < 2; ++s) { // stage W tile 128x32 (fp32 -> bf16)
            int task = t + s * 256;
            int wrow = task >> 2, wseg = task & 3;
            const float4* g = (const float4*)(w + (size_t)(e0 + wrow) * DIM + kt + wseg * 8);
            float4 a = g[0], b = g[1];
            int o = (wrow * 32 + wseg * 8) >> 1;
            sW32[o + 0] = pack_bf16(a.x, a.y);
            sW32[o + 1] = pack_bf16(a.z, a.w);
            sW32[o + 2] = pack_bf16(b.x, b.y);
            sW32[o + 3] = pack_bf16(b.z, b.w);
        }
        if (wave == 0) __builtin_amdgcn_s_wait_tensorcnt(0);
        __syncthreads();

        Frag af[2], bf[2];
#pragma unroll
        for (int s = 0; s < 2; ++s) {
            int arow = wm * 32 + s * 16 + laneN;
            int brow = wn * 32 + s * 16 + laneN;
#pragma unroll
            for (int j = 0; j < 8; ++j) {
                int k = frag_k(j, group);
                af[s].u[j] = sX32[arow * 16 + (k >> 1)];
                bf[s].u[j] = sW32[brow * 16 + (k >> 1)];
            }
        }
#pragma unroll
        for (int i = 0; i < 2; ++i)
#pragma unroll
            for (int j = 0; j < 2; ++j)
                acc[i][j] = __builtin_amdgcn_wmma_f32_16x16x32_bf16(
                    false, af[i].v, false, bf[j].v, (short)0, acc[i][j], false, false);
        __syncthreads();
    }

#pragma unroll
    for (int i = 0; i < 2; ++i)
#pragma unroll
        for (int j = 0; j < 2; ++j) {
            const int mBase = m0 + wm * 32 + i * 16 + (group << 3);
            const int e     = e0 + wn * 32 + j * 16 + laneN;
            const float bv  = bias[e];
#pragma unroll
            for (int r = 0; r < 8; ++r)
                out[(size_t)(mBase + r) * DIM + e] = acc[i][j][r] + bv;
        }
}

// ---------------------------------------------------------------------------
extern "C" void kernel_launch(void* const* d_in, const int* in_sizes, int n_in,
                              void* d_out, int out_size, void* d_ws, size_t ws_size,
                              hipStream_t stream) {
    const float* x     = (const float*)d_in[0];
    const float* w_qkv = (const float*)d_in[1];
    const float* w_out = (const float*)d_in[2];
    const float* b_out = (const float*)d_in[3];
    float* out = (float*)d_out;

    __bf16* ws = (__bf16*)d_ws;
    __bf16* q  = ws + Q_OFF;
    __bf16* kk = ws + K_OFF;
    __bf16* vt = ws + VT_OFF;
    __bf16* xo = ws + XO_OFF;

    // qkv projection: grid = (3072/128, 8192/64)
    qkv_gemm_kernel<<<dim3(24, 128), 256, 0, stream>>>(x, w_qkv, q, kk, vt);
    // attention: 64 (b,h) * 16 row-blocks of 128 rows
    attn_kernel<<<dim3(1024), 256, 0, stream>>>(q, kk, vt, xo);
    // output projection + bias: grid = (1024/128, 8192/64)
    out_gemm_kernel<<<dim3(8, 128), 256, 0, stream>>>(xo, w_out, b_out, out);
}
